// Network_54872502173960
// MI455X (gfx1250) — matrix operand order repaired and checked by reading, not compiled, not run on hardware
//
#include <hip/hip_runtime.h>

#define NDIM   720
#define HALF_N 360
#define BATCH  8
#define NN     (NDIM * NDIM)          // 518400
#define IMGN   (BATCH * NN)           // 4147200
#define BETA_F 100.0f
#define EPS_F  1e-15f
#define PI_F   3.14159265358979323846f

typedef __attribute__((ext_vector_type(16))) _Float16 v16h;
typedef __attribute__((ext_vector_type(8)))  _Float16 v8h;
typedef __attribute__((ext_vector_type(8)))  float    v8f;

#define CAT16(lo, hi) __builtin_shufflevector((lo), (hi), 0,1,2,3,4,5,6,7,8,9,10,11,12,13,14,15)

// ---------------------------------------------------------------------------
// Complex GEMM, DFT-as-matmul workhorse.  C = A * B  (complex, f32 in memory,
// f16 operands, f32 WMMA accumulate).  Per-batch strides sA/sB/sC (0 = shared
// twiddle matrix).  aiScale/biScale conjugate A/B for the inverse transform;
// outScale applies the 1/N per-pass IFFT scaling.
//
// Block: 256 threads = 8 waves; macro tile 64(M) x 32(N); K staged 32 at a
// time in LDS as f16.  A stored [m][k], B stored TRANSPOSED [n][k] so every
// wave fragment is contiguous 16B-aligned runs -> ds_load_b128.  Row stride
// 40 halves keeps 8-element chunks 16B aligned.  Staging is phase-split:
// all coordinates, then ALL global loads (clamped in-bounds, batched so one
// s_wait_loadcnt covers the tile), then convert+mask+ds_store.  No exec-mask
// divergence anywhere.  Each wave: one 16x16 complex tile, 4
// v_wmma_f32_16x16x32_f16 per K step; the Ai*Bi term gets its own
// accumulator so no operand negation is needed.
// ---------------------------------------------------------------------------
#define LDS_A 40
#define LDS_B 40

__global__ void __launch_bounds__(256)
k_cgemm(const float* __restrict__ Ar, const float* __restrict__ Ai, long sA,
        const float* __restrict__ Br, const float* __restrict__ Bi, long sB,
        float* __restrict__ Cr, float* __restrict__ Ci, long sC,
        int M, int Nc, int K, float aiScale, float biScale, float outScale) {
  __shared__ _Float16 sAr[64 * LDS_A];
  __shared__ _Float16 sAi[64 * LDS_A];
  __shared__ _Float16 sBtr[32 * LDS_B];   // transposed: [n][k]
  __shared__ _Float16 sBti[32 * LDS_B];

  const int tid   = threadIdx.x;
  const int lane  = tid & 31;
  const int wave  = tid >> 5;
  const int waveM = wave >> 1;    // 0..3  -> M sub-tile
  const int waveN = wave & 1;     // 0..1  -> N sub-tile
  const int half  = lane >> 4;    // lane half (wave32 WMMA layout)
  const int l16   = lane & 15;
  const int bz    = blockIdx.z;
  const int mTile = blockIdx.y * 64;
  const int nTile = blockIdx.x * 32;

  const float* ArB = Ar + (size_t)bz * sA;
  const float* AiB = Ai ? (Ai + (size_t)bz * sA) : nullptr;
  const float* BrB = Br + (size_t)bz * sB;
  const float* BiB = Bi ? (Bi + (size_t)bz * sB) : nullptr;

  v8f accR  = {};   // sum Ar*Br
  v8f accRn = {};   // sum Ai*Bi   (subtract in epilogue)
  v8f accI  = {};   // sum Ar*Bi + Ai*Br

  for (int k0 = 0; k0 < K; k0 += 32) {
    // ---- phase 0: staging coordinates + masks (clamped, always legal) ----
    size_t pA[4];  bool a0[4], a1[4];  int arw[4], acl[4];
#pragma unroll
    for (int it = 0; it < 4; ++it) {
      int idx = tid + it * 256;
      int r = idx >> 4, c = (idx & 15) * 2;
      int gr = mTile + r;
      int gc = k0 + c;
      bool okr = gr < M;
      a0[it] = okr && (gc     < K);
      a1[it] = okr && (gc + 1 < K);
      int grc = okr ? gr : (M - 1);
      int gcc = (gc + 2 <= K) ? gc : (K - 2);
      pA[it] = (size_t)grc * K + gcc;
      arw[it] = r; acl[it] = c;
    }
    size_t pB0[2], pB1[2];  bool b0[2], b1[2];  int bn[2], bk[2];
#pragma unroll
    for (int it = 0; it < 2; ++it) {
      int idx = tid + it * 256;
      int kk = (idx >> 5) * 2, n = idx & 31;
      int gk = k0 + kk;
      int gc = nTile + n;
      bool okc = gc < Nc;
      b0[it] = okc && (gk     < K);
      b1[it] = okc && (gk + 1 < K);
      int gcc = okc ? gc : (Nc - 1);
      int gk0 = (gk     < K) ? gk     : (K - 1);
      int gk1 = (gk + 1 < K) ? gk + 1 : (K - 1);
      pB0[it] = (size_t)gk0 * Nc + gcc;
      pB1[it] = (size_t)gk1 * Nc + gcc;
      bn[it] = n; bk[it] = kk;
    }

    // ---- phase 1: issue ALL global loads (one wait covers the tile) ----
    float2 lAr[4], lAi[4];
#pragma unroll
    for (int it = 0; it < 4; ++it) lAr[it] = *(const float2*)(ArB + pA[it]);
    if (AiB) {
#pragma unroll
      for (int it = 0; it < 4; ++it) lAi[it] = *(const float2*)(AiB + pA[it]);
    } else {
#pragma unroll
      for (int it = 0; it < 4; ++it) lAi[it] = make_float2(0.f, 0.f);
    }
    float lBr0[2], lBr1[2], lBi0[2], lBi1[2];
#pragma unroll
    for (int it = 0; it < 2; ++it) {
      lBr0[it] = BrB[pB0[it]];
      lBr1[it] = BrB[pB1[it]];
    }
    if (BiB) {
#pragma unroll
      for (int it = 0; it < 2; ++it) {
        lBi0[it] = BiB[pB0[it]];
        lBi1[it] = BiB[pB1[it]];
      }
    } else {
#pragma unroll
      for (int it = 0; it < 2; ++it) { lBi0[it] = 0.f; lBi1[it] = 0.f; }
    }

    {  // speculative prefetch of next K tile -> global_prefetch_b8 (OOB safe)
      int r = tid >> 5, c = tid & 31;
      __builtin_prefetch(ArB + (size_t)(mTile + r) * K + (k0 + 32 + c), 0, 3);
      __builtin_prefetch(BrB + (size_t)(k0 + 32 + r) * Nc + (nTile + c), 0, 3);
    }

    // ---- phase 2: convert + mask + packed LDS stores ----
#pragma unroll
    for (int it = 0; it < 4; ++it) {
      float vr0 = a0[it] ? lAr[it].x : 0.f;
      float vr1 = a1[it] ? lAr[it].y : 0.f;
      float vi0 = a0[it] ? lAi[it].x * aiScale : 0.f;
      float vi1 = a1[it] ? lAi[it].y * aiScale : 0.f;
      union { unsigned u; _Float16 h[2]; } pr, pi;
      pr.h[0] = (_Float16)vr0; pr.h[1] = (_Float16)vr1;
      pi.h[0] = (_Float16)vi0; pi.h[1] = (_Float16)vi1;
      *(unsigned*)&sAr[arw[it] * LDS_A + acl[it]] = pr.u;
      *(unsigned*)&sAi[arw[it] * LDS_A + acl[it]] = pi.u;
    }
#pragma unroll
    for (int it = 0; it < 2; ++it) {
      float vr0 = b0[it] ? lBr0[it] : 0.f;
      float vr1 = b1[it] ? lBr1[it] : 0.f;
      float vi0 = b0[it] ? lBi0[it] * biScale : 0.f;
      float vi1 = b1[it] ? lBi1[it] * biScale : 0.f;
      union { unsigned u; _Float16 h[2]; } pr, pi;
      pr.h[0] = (_Float16)vr0; pr.h[1] = (_Float16)vr1;
      pi.h[0] = (_Float16)vi0; pi.h[1] = (_Float16)vi1;
      *(unsigned*)&sBtr[bn[it] * LDS_B + bk[it]] = pr.u;
      *(unsigned*)&sBti[bn[it] * LDS_B + bk[it]] = pi.u;
    }
    __syncthreads();

    // ---- fragments: contiguous 16B-aligned runs -> ds_load_b128 ----
    // A (16x32 f16): elements 0..15 = K[16h .. 16h+15]
    const _Float16* aR = &sAr [((waveM << 4) + l16) * LDS_A + 16 * half];
    const _Float16* aI = &sAi [((waveM << 4) + l16) * LDS_A + 16 * half];
    // B (32x16 f16, [n][k]): elements 0..15 = K[16h .. 16h+15]
    const _Float16* bR = &sBtr[((waveN << 4) + l16) * LDS_B + 16 * half];
    const _Float16* bI = &sBti[((waveN << 4) + l16) * LDS_B + 16 * half];

    v16h fAr = CAT16(*(const v8h*)aR, *(const v8h*)(aR + 8));
    v16h fAi = CAT16(*(const v8h*)aI, *(const v8h*)(aI + 8));
    v16h fBr = CAT16(*(const v8h*)bR, *(const v8h*)(bR + 8));
    v16h fBi = CAT16(*(const v8h*)bI, *(const v8h*)(bI + 8));

    accR  = __builtin_amdgcn_wmma_f32_16x16x32_f16(false, fAr, false, fBr, (short)0, accR,  false, false);
    accRn = __builtin_amdgcn_wmma_f32_16x16x32_f16(false, fAi, false, fBi, (short)0, accRn, false, false);
    accI  = __builtin_amdgcn_wmma_f32_16x16x32_f16(false, fAr, false, fBi, (short)0, accI,  false, false);
    accI  = __builtin_amdgcn_wmma_f32_16x16x32_f16(false, fAi, false, fBr, (short)0, accI,  false, false);

    __syncthreads();
  }

  // ---- epilogue: f32 C layout -> row = r + 8*half, col = lane%16 ----
  const int col = nTile + (waveN << 4) + l16;
  float* CrB = Cr + (size_t)bz * sC;
  float* CiB = Ci + (size_t)bz * sC;
#pragma unroll
  for (int r = 0; r < 8; ++r) {
    int row = mTile + (waveM << 4) + r + 8 * half;
    if (row < M && col < Nc) {
      size_t p = (size_t)row * Nc + col;
      CrB[p] = (accR[r] - accRn[r]) * outScale;
      CiB[p] = accI[r] * outScale;
    }
  }
}

// ---------------------------------------------------------------------------
// Twiddle matrix:  W[n,k] = exp(-2*pi*i*n*k/N)  (forward; conj applied via
// the +/-1 imag scales in k_cgemm for the inverse transform).
// ---------------------------------------------------------------------------
__global__ void k_gen_dft(float* __restrict__ c, float* __restrict__ s) {
  long i = (long)blockIdx.x * blockDim.x + threadIdx.x;
  if (i >= (long)NN) return;
  int n = (int)(i / NDIM), k = (int)(i % NDIM);
  long long m = ((long long)n * k) % NDIM;
  float ang = -2.f * PI_F * (float)m / (float)NDIM;
  c[i] = cosf(ang);
  s[i] = sinf(ang);
}

// ---------------------------------------------------------------------------
// Elementwise / sampling kernels
// ---------------------------------------------------------------------------
__global__ void k_scale(const float* __restrict__ in, float* __restrict__ o, float s, long n) {
  long i = (long)blockIdx.x * blockDim.x + threadIdx.x;
  if (i < n) o[i] = in[i] * s;
}

__global__ void k_sub_scalar(float* __restrict__ x, const float* __restrict__ m, long n) {
  long i = (long)blockIdx.x * blockDim.x + threadIdx.x;
  if (i < n) x[i] -= m[0];
}

// |F| with fftshift folded in
__global__ void k_magshift(const float* __restrict__ Fr, const float* __restrict__ Fi,
                           float* __restrict__ dst) {
  long i = (long)blockIdx.x * blockDim.x + threadIdx.x;
  if (i >= (long)IMGN) return;
  int b = (int)(i / NN);
  int rem = (int)(i % NN);
  int y = rem / NDIM, x = rem % NDIM;
  int ys = (y + HALF_N) % NDIM, xs = (x + HALF_N) % NDIM;
  size_t p = (size_t)b * NN + (size_t)ys * NDIM + xs;
  dst[i] = sqrtf(Fr[p] * Fr[p] + Fi[p] * Fi[p]);
}

__device__ __forceinline__ float bilin(const float* __restrict__ img, int b, float yy, float xx) {
  float y0f = floorf(yy), x0f = floorf(xx);
  float wy = yy - y0f, wx = xx - x0f;
  float acc = 0.f;
#pragma unroll
  for (int dy = 0; dy < 2; ++dy) {
#pragma unroll
    for (int dx = 0; dx < 2; ++dx) {
      float yi = y0f + dy, xi = x0f + dx;
      bool valid = (yi >= 0.f) && (yi <= (float)(NDIM - 1)) &&
                   (xi >= 0.f) && (xi <= (float)(NDIM - 1));
      int yc = min(max((int)yi, 0), NDIM - 1);
      int xc = min(max((int)xi, 0), NDIM - 1);
      float v = valid ? img[(size_t)b * NN + (size_t)yc * NDIM + xc] : 0.f;
      float w = (dy ? wy : 1.f - wy) * (dx ? wx : 1.f - wx);
      acc += v * w;
    }
  }
  return acc;
}

// log-polar resample of the shifted magnitude + high-pass row weight
__global__ void k_logpolar(const float* __restrict__ mag, float* __restrict__ dst) {
  long i = (long)blockIdx.x * blockDim.x + threadIdx.x;
  if (i >= (long)IMGN) return;
  int b = (int)(i / NN);
  int rem = (int)(i % NN);
  int y = rem / NDIM, x = rem % NDIM;
  float log_base = logf((float)NDIM / 4.f) / (float)NDIM;   // log(180)/720
  float radius = expf((float)y * log_base);
  float theta = (float)x * (2.f * PI_F / (float)NDIM);
  float yy = (float)HALF_N + radius * sinf(theta);
  float xx = (float)HALF_N + radius * cosf(theta);
  float t = (float)y * (5.f / 719.f);                       // linspace(0,5,720)
  float w = t * t; w = w * w * t;                           // t^5
  dst[i] = bilin(mag, b, yy, xx) * w;
}

// cross power spectral density, in place into (Ar,Ai)
__global__ void k_cpsd(float* __restrict__ Ar, float* __restrict__ Ai,
                       const float* __restrict__ Br, const float* __restrict__ Bi) {
  long i = (long)blockIdx.x * blockDim.x + threadIdx.x;
  if (i >= (long)IMGN) return;
  float ar = Ar[i], ai = Ai[i], br = Br[i], bi = Bi[i];
  float d = sqrtf(ar * ar + ai * ai) * sqrtf(br * br + bi * bi) + EPS_F;
  Ar[i] = (ar * br + ai * bi) / d;
  Ai[i] = (ai * br - ar * bi) / d;
}

// fftshifted real part of the correlation surface
__global__ void k_shiftreal(const float* __restrict__ Sr, float* __restrict__ dst) {
  long i = (long)blockIdx.x * blockDim.x + threadIdx.x;
  if (i >= (long)IMGN) return;
  int b = (int)(i / NN);
  int rem = (int)(i % NN);
  int y = rem / NDIM, x = rem % NDIM;
  int ys = (y + HALF_N) % NDIM, xs = (x + HALF_N) % NDIM;
  dst[i] = Sr[(size_t)b * NN + (size_t)ys * NDIM + xs];
}

// affine warp (rotation+scale, zero translation), bilinear, zero pad
__global__ void k_warp(const float* __restrict__ src, const float* __restrict__ mat,
                       float* __restrict__ dst) {
  long i = (long)blockIdx.x * blockDim.x + threadIdx.x;
  if (i >= (long)IMGN) return;
  int b = (int)(i / NN);
  int rem = (int)(i % NN);
  int y = rem / NDIM, x = rem % NDIM;
  float c = mat[2 * b], s = mat[2 * b + 1];
  float xn = (2.f * x + 1.f) / (float)NDIM - 1.f;
  float yn = (2.f * y + 1.f) / (float)NDIM - 1.f;
  float sx = c * xn - s * yn;
  float sy = s * xn + c * yn;
  float px = ((sx + 1.f) * (float)NDIM - 1.f) * 0.5f;
  float py = ((sy + 1.f) * (float)NDIM - 1.f) * 0.5f;
  dst[i] = bilin(src, b, py, px);
}

// ---------------------------------------------------------------------------
// Deterministic two-stage reductions
// ---------------------------------------------------------------------------
__global__ void k_sum_part(const float* __restrict__ x, long n, float* __restrict__ part) {
  __shared__ float sm[256];
  float s = 0.f;
  for (long i = (long)blockIdx.x * blockDim.x + threadIdx.x; i < n;
       i += (long)gridDim.x * blockDim.x) s += x[i];
  sm[threadIdx.x] = s; __syncthreads();
  for (int o = 128; o > 0; o >>= 1) {
    if ((int)threadIdx.x < o) sm[threadIdx.x] += sm[threadIdx.x + o];
    __syncthreads();
  }
  if (!threadIdx.x) part[blockIdx.x] = sm[0];
}

__global__ void k_entr_part(const float* __restrict__ x, long n, float* __restrict__ part) {
  __shared__ float sm[256];
  float s = 0.f;
  for (long i = (long)blockIdx.x * blockDim.x + threadIdx.x; i < n;
       i += (long)gridDim.x * blockDim.x) {
    float v = fabsf(x[i]);
    s += (v > 0.f) ? (-v * logf(v)) : 0.f;
  }
  sm[threadIdx.x] = s; __syncthreads();
  for (int o = 128; o > 0; o >>= 1) {
    if ((int)threadIdx.x < o) sm[threadIdx.x] += sm[threadIdx.x + o];
    __syncthreads();
  }
  if (!threadIdx.x) part[blockIdx.x] = sm[0];
}

__global__ void k_sum_fin(const float* __restrict__ part, int np, float* __restrict__ dst,
                          float scale) {
  __shared__ float sm[256];
  float s = 0.f;
  for (int i = threadIdx.x; i < np; i += 256) s += part[i];
  sm[threadIdx.x] = s; __syncthreads();
  for (int o = 128; o > 0; o >>= 1) {
    if ((int)threadIdx.x < o) sm[threadIdx.x] += sm[threadIdx.x + o];
    __syncthreads();
  }
  if (!threadIdx.x) dst[0] = sm[0] * scale;
}

// per-batch max for stable softmax
__global__ void k_max_part(const float* __restrict__ x, float* __restrict__ part) {
  __shared__ float sm[256];
  int b = blockIdx.y;
  const float* xb = x + (size_t)b * NN;
  float m = -3.4e38f;
  for (long i = (long)blockIdx.x * blockDim.x + threadIdx.x; i < (long)NN;
       i += (long)gridDim.x * blockDim.x) m = fmaxf(m, xb[i]);
  sm[threadIdx.x] = m; __syncthreads();
  for (int o = 128; o > 0; o >>= 1) {
    if ((int)threadIdx.x < o) sm[threadIdx.x] = fmaxf(sm[threadIdx.x], sm[threadIdx.x + o]);
    __syncthreads();
  }
  if (!threadIdx.x) part[(size_t)b * gridDim.x + blockIdx.x] = sm[0];
}

__global__ void k_max_fin(const float* __restrict__ part, int np, float* __restrict__ bmax) {
  __shared__ float sm[128];
  int b = blockIdx.x;
  float m = -3.4e38f;
  for (int i = threadIdx.x; i < np; i += 128) m = fmaxf(m, part[(size_t)b * np + i]);
  sm[threadIdx.x] = m; __syncthreads();
  for (int o = 64; o > 0; o >>= 1) {
    if ((int)threadIdx.x < o) sm[threadIdx.x] = fmaxf(sm[threadIdx.x], sm[threadIdx.x + o]);
    __syncthreads();
  }
  if (!threadIdx.x) bmax[b] = sm[0];
}

// per-batch softmax moment partials: sum(e), sum(e*y), sum(e*x)
__global__ void k_sm_part(const float* __restrict__ x, const float* __restrict__ bmax,
                          float* __restrict__ part3) {
  __shared__ float se[256], syh[256], sxh[256];
  int b = blockIdx.y;
  const float* xb = x + (size_t)b * NN;
  float mx = bmax[b];
  float ae = 0.f, ay = 0.f, ax = 0.f;
  for (long i = (long)blockIdx.x * blockDim.x + threadIdx.x; i < (long)NN;
       i += (long)gridDim.x * blockDim.x) {
    float e = expf(BETA_F * (xb[i] - mx));
    int yy = (int)(i / NDIM), xx = (int)(i % NDIM);
    ae += e; ay += e * (float)yy; ax += e * (float)xx;
  }
  se[threadIdx.x] = ae; syh[threadIdx.x] = ay; sxh[threadIdx.x] = ax;
  __syncthreads();
  for (int o = 128; o > 0; o >>= 1) {
    if ((int)threadIdx.x < o) {
      se[threadIdx.x] += se[threadIdx.x + o];
      syh[threadIdx.x] += syh[threadIdx.x + o];
      sxh[threadIdx.x] += sxh[threadIdx.x + o];
    }
    __syncthreads();
  }
  if (!threadIdx.x) {
    size_t s = ((size_t)b * gridDim.x + blockIdx.x) * 3;
    part3[s] = se[0]; part3[s + 1] = syh[0]; part3[s + 2] = sxh[0];
  }
}

__global__ void k_sm_fin(const float* __restrict__ part3, int np, float* __restrict__ shiftv) {
  __shared__ float se[128], syh[128], sxh[128];
  int b = blockIdx.x;
  float ae = 0.f, ay = 0.f, ax = 0.f;
  for (int i = threadIdx.x; i < np; i += 128) {
    size_t s = ((size_t)b * np + i) * 3;
    ae += part3[s]; ay += part3[s + 1]; ax += part3[s + 2];
  }
  se[threadIdx.x] = ae; syh[threadIdx.x] = ay; sxh[threadIdx.x] = ax;
  __syncthreads();
  for (int o = 64; o > 0; o >>= 1) {
    if ((int)threadIdx.x < o) {
      se[threadIdx.x] += se[threadIdx.x + o];
      syh[threadIdx.x] += syh[threadIdx.x + o];
      sxh[threadIdx.x] += sxh[threadIdx.x + o];
    }
    __syncthreads();
  }
  if (!threadIdx.x) {
    float S = se[0];
    float ey = syh[0] / S, ex = sxh[0] / S;
    shiftv[2 * b]     = (ey - (float)HALF_N) / (float)NDIM;
    shiftv[2 * b + 1] = (ex - (float)HALF_N) / (float)NDIM;
  }
}

// rotation / scale / affine matrices; writes out[:,0:2]
__global__ void k_rotscale(const float* __restrict__ shiftv, float* __restrict__ matM,
                           float* __restrict__ matMt, float* __restrict__ outp) {
  int b = threadIdx.x;
  if (b >= BATCH) return;
  float rot = -shiftv[2 * b + 1] * 360.f;
  float scl = expf(logf(180.f) * shiftv[2 * b]);     // r = log(max(cy/2, cx/2))
  float rad = rot * (PI_F / 180.f);
  matM[2 * b]     = cosf(rad) * scl;
  matM[2 * b + 1] = sinf(rad) * scl;
  float radt = (rot - 180.f) * (PI_F / 180.f);
  matMt[2 * b]     = cosf(radt) * scl;
  matMt[2 * b + 1] = sinf(radt) * scl;
  outp[4 * b]     = rot;
  outp[4 * b + 1] = scl;
}

// entropy-based selection; writes out[:,2:4]
__global__ void k_final(const float* __restrict__ e2, const float* __restrict__ sh2,
                        const float* __restrict__ sh2t, float* __restrict__ outp) {
  int b = threadIdx.x;
  if (b >= BATCH) return;
  bool sel = e2[0] < e2[1];
  outp[4 * b + 2] = sel ? sh2[2 * b]     : sh2t[2 * b];
  outp[4 * b + 3] = sel ? sh2[2 * b + 1] : sh2t[2 * b + 1];
}

// ---------------------------------------------------------------------------
// Host driver
// ---------------------------------------------------------------------------
extern "C" void kernel_launch(void* const* d_in, const int* in_sizes, int n_in,
                              void* d_out, int out_size, void* d_ws, size_t ws_size,
                              hipStream_t stream) {
  const float* img_o = (const float*)d_in[0];
  const float* tem_o = (const float*)d_in[1];
  float* out = (float*)d_out;
  float* cr1 = out + 32;            // (8,1,720,720)
  float* cr2 = cr1 + IMGN;          // (8,1,720,720)

  float* ws = (float*)d_ws;
  float* img  = ws + 0L  * IMGN;
  float* tem  = ws + 1L  * IMGN;
  float* r_i  = ws + 2L  * IMGN;
  float* r_it = ws + 3L  * IMGN;
  float* imf  = ws + 4L  * IMGN;    // reused later as cr2_t
  float* tef  = ws + 5L  * IMGN;
  float* S0   = ws + 6L  * IMGN;
  float* S1   = ws + 7L  * IMGN;
  float* S2   = ws + 8L  * IMGN;
  float* S3   = ws + 9L  * IMGN;
  float* S4   = ws + 10L * IMGN;
  float* S5   = ws + 11L * IMGN;
  float* dftc = ws + 12L * IMGN;
  float* dfts = dftc + NN;
  float* smallR = dfts + NN;
  float* part    = smallR + 0;      // 512
  float* meanS   = smallR + 512;
  float* eSlots  = smallR + 520;    // 2
  float* bmax    = smallR + 528;    // 8
  float* shift1  = smallR + 536;    // 16
  float* shift2  = smallR + 552;    // 16
  float* shift2t = smallR + 568;    // 16
  float* matM    = smallR + 584;    // 16
  float* matMt   = smallR + 600;    // 16
  float* maxPart = smallR + 640;    // 8*128
  float* smPart  = smallR + 1664;   // 8*128*3

  const dim3 G1((IMGN + 255) / 256);
  const dim3 GW(256);

  auto cgemm = [&](const float* Ar_, const float* Ai_, long sA_,
                   const float* Br_, const float* Bi_, long sB_,
                   float* Cr_, float* Ci_, long sC_,
                   float aiS, float biS, float oS) {
    dim3 g((NDIM + 31) / 32, (NDIM + 63) / 64, BATCH);
    k_cgemm<<<g, GW, 0, stream>>>(Ar_, Ai_, sA_, Br_, Bi_, sB_, Cr_, Ci_, sC_,
                                  NDIM, NDIM, NDIM, aiS, biS, oS);
  };
  // fft2: F = W * (X * W); inverse uses conj(W) and 1/N per pass
  auto fft2 = [&](const float* Xr, const float* Xi, float* Fr, float* Fi, bool inv) {
    float s  = inv ? -1.f : 1.f;
    float os = inv ? (1.f / (float)NDIM) : 1.f;
    cgemm(Xr, Xi, NN, dftc, dfts, 0, S4, S5, NN, 1.f, s, os);   // row pass
    cgemm(dftc, dfts, 0, S4, S5, NN, Fr, Fi, NN, s, 1.f, os);   // col pass
  };
  auto meansub = [&](float* x) {
    k_sum_part<<<512, 256, 0, stream>>>(x, (long)IMGN, part);
    k_sum_fin<<<1, 256, 0, stream>>>(part, 512, meanS, 1.f / (float)IMGN);
    k_sub_scalar<<<G1, 256, 0, stream>>>(x, meanS, (long)IMGN);
  };
  auto calc_shift = [&](float* A_, float* B_, float* crDst, float* shiftDst) {
    meansub(A_); meansub(A_);        // reference subtracts img mean twice
    meansub(B_);
    fft2(A_, nullptr, S0, S1, false);
    fft2(B_, nullptr, S2, S3, false);
    k_cpsd<<<G1, 256, 0, stream>>>(S0, S1, S2, S3);
    // ifft2 of (S0,S1) -> (S2,S3)
    cgemm(S0, S1, NN, dftc, dfts, 0, S4, S5, NN, 1.f, -1.f, 1.f / (float)NDIM);
    cgemm(dftc, dfts, 0, S4, S5, NN, S2, S3, NN, -1.f, 1.f, 1.f / (float)NDIM);
    k_shiftreal<<<G1, 256, 0, stream>>>(S2, crDst);
    k_max_part<<<dim3(128, BATCH), 256, 0, stream>>>(crDst, maxPart);
    k_max_fin<<<BATCH, 128, 0, stream>>>(maxPart, 128, bmax);
    k_sm_part<<<dim3(128, BATCH), 256, 0, stream>>>(crDst, bmax, smPart);
    k_sm_fin<<<BATCH, 128, 0, stream>>>(smPart, 128, shiftDst);
  };

  // --- pipeline ---
  k_gen_dft<<<(NN + 255) / 256, 256, 0, stream>>>(dftc, dfts);
  k_scale<<<G1, 256, 0, stream>>>(img_o, img, 1.f / 255.f, (long)IMGN);
  k_scale<<<G1, 256, 0, stream>>>(tem_o, tem, 1.f / 255.f, (long)IMGN);

  // spectral magnitudes
  fft2(img, nullptr, S0, S1, false);
  k_magshift<<<G1, 256, 0, stream>>>(S0, S1, S2);
  fft2(tem, nullptr, S0, S1, false);
  k_magshift<<<G1, 256, 0, stream>>>(S0, S1, S3);

  // log-polar + weight
  k_logpolar<<<G1, 256, 0, stream>>>(S2, imf);
  k_logpolar<<<G1, 256, 0, stream>>>(S3, tef);

  // rotation / scale
  calc_shift(imf, tef, cr1, shift1);
  k_rotscale<<<1, 32, 0, stream>>>(shift1, matM, matMt, out);

  // warps
  k_warp<<<G1, 256, 0, stream>>>(tem, matM, r_i);
  k_warp<<<G1, 256, 0, stream>>>(tem, matMt, r_it);

  // translation (two hypotheses)
  calc_shift(img, r_i, cr2, shift2);
  calc_shift(img, r_it, imf /* cr2_t */, shift2t);

  // entropy select
  k_entr_part<<<512, 256, 0, stream>>>(cr2, (long)IMGN, part);
  k_sum_fin<<<1, 256, 0, stream>>>(part, 512, eSlots + 0, 1.f / (float)IMGN);
  k_entr_part<<<512, 256, 0, stream>>>(imf, (long)IMGN, part);
  k_sum_fin<<<1, 256, 0, stream>>>(part, 512, eSlots + 1, 1.f / (float)IMGN);
  k_final<<<1, 32, 0, stream>>>(eSlots, shift2, shift2t, out);
}
